// KermtDualCausalCv4_68702296867109
// MI455X (gfx1250) — matrix operand order repaired and checked by reading, not compiled, not run on hardware
//
#include <hip/hip_runtime.h>
#include <hip/hip_bf16.h>
#include <math.h>

typedef __attribute__((ext_vector_type(16))) _Float16 v16h;
typedef __attribute__((ext_vector_type(8)))  _Float16 v8h;
typedef __attribute__((ext_vector_type(8)))  float    v8f;

#define TSTEPS 4096
#define HSZ    2048
#define NSOLV  5
#define NDESC  6
#define DBLK   (HSZ + NDESC)         // 2054
#define DIN    (DBLK*(1+NSOLV) + 1)  // 12325

__device__ __forceinline__ float gelu_exact(float x) {
    return 0.5f * x * (1.0f + erff(x * 0.70710678118654752f));
}

// ---------------- f32 -> f16 weight conversion ----------------
__global__ void k_f32_to_f16(const float* __restrict__ src,
                             _Float16* __restrict__ dst, int n) {
    int i = blockIdx.x * blockDim.x + threadIdx.x;
    if (i < n) dst[i] = (_Float16)src[i];
}

// ---------------- factorized layer-0 precompute ----------------
// One wave per output unit h. Computes:
//   base[h]  = b0[h] + dot(W0[h, 0:2048], mol)
//   V[s][h]  = dot(W0[h, sol_s block], solvent_vecs[s])
//   Wd[h][d] = sum over the 6 desc column-blocks of W0[h, .]
//   wlast[h] = W0[h, DIN-1]
__global__ void k_precompute(const float* __restrict__ W0,
                             const float* __restrict__ b0,
                             const float* __restrict__ mol,
                             const float* __restrict__ solv,   // [NSOLV, HSZ]
                             float* __restrict__ base,         // [HSZ]
                             float* __restrict__ V,            // [NSOLV*HSZ]
                             float* __restrict__ Wd,           // [HSZ*NDESC]
                             float* __restrict__ wlast) {      // [HSZ]
    const int wave = threadIdx.x >> 5;
    const int lane = threadIdx.x & 31;
    const int h = blockIdx.x * (blockDim.x >> 5) + wave;
    if (h >= HSZ) return;
    const float* row = W0 + (size_t)h * DIN;
    float sm = 0.f, s0 = 0.f, s1 = 0.f, s2 = 0.f, s3 = 0.f, s4 = 0.f;
    for (int j = lane; j < HSZ; j += 32) {
        sm += row[j]            * mol[j];
        s0 += row[DBLK*1 + j]   * solv[0*HSZ + j];
        s1 += row[DBLK*2 + j]   * solv[1*HSZ + j];
        s2 += row[DBLK*3 + j]   * solv[2*HSZ + j];
        s3 += row[DBLK*4 + j]   * solv[3*HSZ + j];
        s4 += row[DBLK*5 + j]   * solv[4*HSZ + j];
    }
    for (int off = 16; off > 0; off >>= 1) {
        sm += __shfl_xor(sm, off, 32);
        s0 += __shfl_xor(s0, off, 32);
        s1 += __shfl_xor(s1, off, 32);
        s2 += __shfl_xor(s2, off, 32);
        s3 += __shfl_xor(s3, off, 32);
        s4 += __shfl_xor(s4, off, 32);
    }
    if (lane == 0) {
        base[h] = b0[h] + sm;
        V[0*HSZ + h] = s0;
        V[1*HSZ + h] = s1;
        V[2*HSZ + h] = s2;
        V[3*HSZ + h] = s3;
        V[4*HSZ + h] = s4;
        wlast[h] = row[DIN - 1];
        #pragma unroll
        for (int d = 0; d < NDESC; ++d) {
            float acc = row[HSZ + d];
            #pragma unroll
            for (int s = 1; s <= NSOLV; ++s) acc += row[DBLK*s + HSZ + d];
            Wd[h*NDESC + d] = acc;
        }
    }
}

// ---------------- build h0 = gelu(layer0) as f16 ----------------
__global__ void k_build_h0(const float* __restrict__ base,
                           const float* __restrict__ V,
                           const float* __restrict__ Wd,
                           const float* __restrict__ wlast,
                           const float* __restrict__ sseq,   // [T, NSOLV]
                           const float* __restrict__ desc,   // [T, NDESC]
                           const float* __restrict__ rf,     // [T]
                           int is_fwd,
                           _Float16* __restrict__ out) {     // [T, HSZ] f16
    const int h = blockIdx.x * blockDim.x + threadIdx.x;
    const int t = blockIdx.y;
    float pn;
    if (is_fwd) pn = (t == 0)          ? 0.5f : rf[t - 1];
    else        pn = (t == TSTEPS - 1) ? 0.5f : rf[t + 1];
    float acc = base[h] + pn * wlast[h];
    #pragma unroll
    for (int s = 0; s < NSOLV; ++s) acc += sseq[t*NSOLV + s] * V[s*HSZ + h];
    #pragma unroll
    for (int d = 0; d < NDESC; ++d) acc += desc[t*NDESC + d] * Wd[h*NDESC + d];
    out[(size_t)t * HSZ + h] = (_Float16)gelu_exact(acc);
}

// ---------------- WMMA GEMM: C = gelu(A * B^T + bias), f16 in, f32 acc --------
// A: [M=4096, K=2048] row-major f16 (activations)
// B: [N=2048, K=2048] row-major f16 (weights [out, in] -> C = A * B^T)
// Block = 256 threads = 8 waves in a 2(M) x 4(N) grid of 64x64 wave tiles
// -> 128 x 256 block tile. A/B K-slices staged through double-buffered LDS so
// each slice is read from L2 exactly once per block (~87 FLOP / L2 byte).
#define BM 128
#define BN 256
#define BK 32
#define LDA_S 40   // 80B row stride: 80*r mod 256 covers all 16B offsets -> no conflicts
#define LDB_S 40

__global__ void __launch_bounds__(256)
k_wmma_gemm_bias_gelu(const _Float16* __restrict__ A,
                      const _Float16* __restrict__ B,
                      const float*    __restrict__ bias,
                      _Float16*       __restrict__ C) {
    constexpr int N = HSZ, K = HSZ;
    constexpr int NK = K / BK;                       // 64 K-steps

    __shared__ _Float16 sA[2][BM][LDA_S];
    __shared__ _Float16 sB[2][BN][LDB_S];

    const int tid  = threadIdx.x;
    const int wave = tid >> 5;
    const int lane = tid & 31;
    const int wr   = wave >> 2;                      // 0..1  (M)
    const int wc   = wave & 3;                       // 0..3  (N)
    const int m0   = blockIdx.y * BM;
    const int n0   = blockIdx.x * BN;
    const int half = lane >> 4;                      // 0 or 1
    const int l16  = lane & 15;

    // cooperative-load assignments (全 256 threads)
    const int arow = tid >> 1;                       // 0..127
    const int aseg = tid & 1;                        // 0..1 (16-half segment)
    const int brow = tid;                            // 0..255

    const _Float16* gA = A + (size_t)(m0 + arow) * K + aseg * 16;
    const _Float16* gB = B + (size_t)(n0 + brow) * K;

    union V16 { v16h v; v8h h8[2]; };

    const v8f vzero = {0.f, 0.f, 0.f, 0.f, 0.f, 0.f, 0.f, 0.f};
    v8f acc[4][4];
    #pragma unroll
    for (int i = 0; i < 4; ++i)
        #pragma unroll
        for (int j = 0; j < 4; ++j)
            acc[i][j] = vzero;

    // prologue: stage k-slice 0
    V16 asg, bsg0, bsg1;
    asg.v  = *(const v16h*)(gA);
    bsg0.v = *(const v16h*)(gB);
    bsg1.v = *(const v16h*)(gB + 16);
    *(v8h*)&sA[0][arow][aseg*16    ] = asg.h8[0];
    *(v8h*)&sA[0][arow][aseg*16 + 8] = asg.h8[1];
    *(v8h*)&sB[0][brow][0 ]          = bsg0.h8[0];
    *(v8h*)&sB[0][brow][8 ]          = bsg0.h8[1];
    *(v8h*)&sB[0][brow][16]          = bsg1.h8[0];
    *(v8h*)&sB[0][brow][24]          = bsg1.h8[1];
    __syncthreads();

    for (int kt = 0; kt < NK; ++kt) {
        const int buf = kt & 1;
        // prefetch next k-slice into registers (overlaps with WMMAs below)
        if (kt + 1 < NK) {
            const int ko = (kt + 1) * BK;
            asg.v  = *(const v16h*)(gA + ko);
            bsg0.v = *(const v16h*)(gB + ko);
            bsg1.v = *(const v16h*)(gB + ko + 16);
        }

        // load fragments from LDS and issue 16 WMMAs
        v16h af[4], bf[4];
        #pragma unroll
        for (int i = 0; i < 4; ++i) {
            const int row = wr*64 + 16*i + l16;
            V16 u;
            u.h8[0] = *(const v8h*)&sA[buf][row][half*8];
            u.h8[1] = *(const v8h*)&sA[buf][row][16 + half*8];
            af[i] = u.v;
        }
        #pragma unroll
        for (int j = 0; j < 4; ++j) {
            const int row = wc*64 + 16*j + l16;
            V16 u;
            u.h8[0] = *(const v8h*)&sB[buf][row][half*8];
            u.h8[1] = *(const v8h*)&sB[buf][row][16 + half*8];
            bf[j] = u.v;
        }
        #pragma unroll
        for (int i = 0; i < 4; ++i)
            #pragma unroll
            for (int j = 0; j < 4; ++j)
                acc[i][j] = __builtin_amdgcn_wmma_f32_16x16x32_f16(
                    false, af[i], false, bf[j], (short)0, acc[i][j], false, false);

        __syncthreads();                 // all reads of the other buffer done
        if (kt + 1 < NK) {
            const int nb = buf ^ 1;
            *(v8h*)&sA[nb][arow][aseg*16    ] = asg.h8[0];
            *(v8h*)&sA[nb][arow][aseg*16 + 8] = asg.h8[1];
            *(v8h*)&sB[nb][brow][0 ]          = bsg0.h8[0];
            *(v8h*)&sB[nb][brow][8 ]          = bsg0.h8[1];
            *(v8h*)&sB[nb][brow][16]          = bsg1.h8[0];
            *(v8h*)&sB[nb][brow][24]          = bsg1.h8[1];
            __syncthreads();             // new buffer visible to all waves
        }
    }

    // Epilogue: D layout — lane n = l16, VGPR r holds M = r + 8*half.
    #pragma unroll
    for (int j = 0; j < 4; ++j) {
        const int n = n0 + wc*64 + 16*j + l16;
        const float bn = bias[n];
        #pragma unroll
        for (int i = 0; i < 4; ++i) {
            #pragma unroll
            for (int r = 0; r < 8; ++r) {
                const int m = m0 + wr*64 + 16*i + half*8 + r;
                C[(size_t)m * N + n] = (_Float16)gelu_exact(acc[i][j][r] + bn);
            }
        }
    }
}

// ---------------- output head: out[t] = clip(h2[t,:].Wo + bo) ----------------
__global__ void k_head(const _Float16* __restrict__ h2,
                       const float* __restrict__ Wo,
                       const float* __restrict__ bo,
                       float* __restrict__ out) {
    __shared__ float red[256];
    const int t = blockIdx.x;
    float s = 0.f;
    for (int hh = threadIdx.x; hh < HSZ; hh += 256)
        s += (float)h2[(size_t)t * HSZ + hh] * Wo[hh];
    red[threadIdx.x] = s;
    __syncthreads();
    for (int off = 128; off > 0; off >>= 1) {
        if (threadIdx.x < (unsigned)off) red[threadIdx.x] += red[threadIdx.x + off];
        __syncthreads();
    }
    if (threadIdx.x == 0) {
        float v = red[0] + bo[0];
        v = fminf(fmaxf(v, 1e-4f), 1.0f - 1e-4f);
        out[t] = v;
    }
}

extern "C" void kernel_launch(void* const* d_in, const int* in_sizes, int n_in,
                              void* d_out, int out_size, void* d_ws, size_t ws_size,
                              hipStream_t stream) {
    (void)in_sizes; (void)n_in; (void)out_size; (void)ws_size;
    // Inputs, in setup_inputs() order
    const float* mol   = (const float*)d_in[0];
    const float* sseq  = (const float*)d_in[1];
    const float* desc  = (const float*)d_in[2];
    const float* rf    = (const float*)d_in[3];
    const float* solv  = (const float*)d_in[4];
    const float* Wf0 = (const float*)d_in[5];
    const float* Wf1 = (const float*)d_in[6];
    const float* Wf2 = (const float*)d_in[7];
    const float* Wof = (const float*)d_in[8];
    const float* Wb0 = (const float*)d_in[9];
    const float* Wb1 = (const float*)d_in[10];
    const float* Wb2 = (const float*)d_in[11];
    const float* Wob = (const float*)d_in[12];
    const float* bf0 = (const float*)d_in[13];
    const float* bf1 = (const float*)d_in[14];
    const float* bf2 = (const float*)d_in[15];
    const float* bof = (const float*)d_in[16];
    const float* bb0 = (const float*)d_in[17];
    const float* bb1 = (const float*)d_in[18];
    const float* bb2 = (const float*)d_in[19];
    const float* bob = (const float*)d_in[20];
    float* out = (float*)d_out;

    // Workspace carve-up
    char* ws = (char*)d_ws;
    _Float16* wf1h = (_Float16*)(ws + ((size_t)0  << 20));  // 8 MB each
    _Float16* wf2h = (_Float16*)(ws + ((size_t)8  << 20));
    _Float16* wb1h = (_Float16*)(ws + ((size_t)16 << 20));
    _Float16* wb2h = (_Float16*)(ws + ((size_t)24 << 20));
    _Float16* bufA = (_Float16*)(ws + ((size_t)32 << 20));  // 16 MB
    _Float16* bufB = (_Float16*)(ws + ((size_t)48 << 20));  // 16 MB
    float* preF = (float*)(ws + ((size_t)64 << 20));        // 13*HSZ floats
    float* preB = preF + 13 * HSZ;
    // per-trunk precompute layout: base | V(5H) | Wd(6H) | wlast
    float *baseF = preF,            *VF = preF + HSZ;
    float *WdF   = preF + 6 * HSZ,  *wlF = preF + 12 * HSZ;
    float *baseB = preB,            *VB = preB + HSZ;
    float *WdB   = preB + 6 * HSZ,  *wlB = preB + 12 * HSZ;

    const int NW = HSZ * HSZ;                   // 2048*2048
    const dim3 cvt_g((NW + 255) / 256), th(256);
    k_f32_to_f16<<<cvt_g, th, 0, stream>>>(Wf1, wf1h, NW);
    k_f32_to_f16<<<cvt_g, th, 0, stream>>>(Wf2, wf2h, NW);
    k_f32_to_f16<<<cvt_g, th, 0, stream>>>(Wb1, wb1h, NW);
    k_f32_to_f16<<<cvt_g, th, 0, stream>>>(Wb2, wb2h, NW);

    // 8 waves/block, one wave per h -> 256 blocks cover HSZ
    k_precompute<<<HSZ / 8, 256, 0, stream>>>(Wf0, bf0, mol, solv, baseF, VF, WdF, wlF);
    k_precompute<<<HSZ / 8, 256, 0, stream>>>(Wb0, bb0, mol, solv, baseB, VB, WdB, wlB);

    const dim3 h0grid(HSZ / 256, TSTEPS);
    const dim3 ggrid(HSZ / BN, TSTEPS / BM);    // (8, 32) blocks of 128x256

    // forward trunk
    k_build_h0<<<h0grid, 256, 0, stream>>>(baseF, VF, WdF, wlF, sseq, desc, rf, 1, bufA);
    k_wmma_gemm_bias_gelu<<<ggrid, 256, 0, stream>>>(bufA, wf1h, bf1, bufB);
    k_wmma_gemm_bias_gelu<<<ggrid, 256, 0, stream>>>(bufB, wf2h, bf2, bufA);
    k_head<<<TSTEPS, 256, 0, stream>>>(bufA, Wof, bof, out);

    // backward trunk
    k_build_h0<<<h0grid, 256, 0, stream>>>(baseB, VB, WdB, wlB, sseq, desc, rf, 0, bufA);
    k_wmma_gemm_bias_gelu<<<ggrid, 256, 0, stream>>>(bufA, wb1h, bb1, bufB);
    k_wmma_gemm_bias_gelu<<<ggrid, 256, 0, stream>>>(bufB, wb2h, bb2, bufA);
    k_head<<<TSTEPS, 256, 0, stream>>>(bufA, Wob, bob, out + TSTEPS);
}
